// RecursiveLSTM_59768764891148
// MI455X (gfx1250) — matrix operand order repaired and checked by reading, not compiled, not run on hardware
//
#include <hip/hip_runtime.h>
#include <hip/hip_bf16.h>

typedef __attribute__((ext_vector_type(16))) __bf16 v16bf;
typedef __attribute__((ext_vector_type(8)))  __bf16 v8bf;
typedef __attribute__((ext_vector_type(8)))  float  v8f;

#define B_   32
#define E_   512
#define H_   1024
#define V_   32000
#define T_   48

#define SA_STRIDE (E_ + H_ + 8)   // 1544: row stride in bf16 elems, +8 pad => 4-bank row skew
#define SH_STRIDE (H_ + 8)        // 1032: same trick for logits A staging

__device__ __forceinline__ unsigned short f2bf(float f) {
    unsigned u = __float_as_uint(f);
    u += 0x7FFFu + ((u >> 16) & 1u);           // round-to-nearest-even
    return (unsigned short)(u >> 16);
}
__device__ __forceinline__ float sigf(float x) { return 1.0f / (1.0f + __expf(-x)); }

// ---------------- one-time weight conversion f32 -> bf16 ----------------
__global__ void cvt_bf16(const float* __restrict__ src, unsigned short* __restrict__ dst, int n) {
    int i = blockIdx.x * blockDim.x + threadIdx.x;
    int stride = gridDim.x * blockDim.x;
    for (; i < n; i += stride) dst[i] = f2bf(src[i]);
}

// ---------------- h0 = cond @ W_h.T + b_h ; c0 likewise ----------------
__global__ void __launch_bounds__(256)
init_state(const float* __restrict__ cond,
           const float* __restrict__ Wh, const float* __restrict__ bh,
           const float* __restrict__ Wc, const float* __restrict__ bc,
           float* __restrict__ h0, float* __restrict__ c0,
           unsigned short* __restrict__ hbf0, int* __restrict__ xtok) {
    const int idx = blockIdx.x * blockDim.x + threadIdx.x;   // 0..32767
    const int b = idx >> 10, n = idx & (H_ - 1);
    const float* cp  = cond + (size_t)b * E_;
    const float* whp = Wh + (size_t)n * E_;
    const float* wcp = Wc + (size_t)n * E_;
    float ah = 0.f, ac = 0.f;
    for (int k = 0; k < E_; ++k) {
        const float cv = cp[k];
        ah = fmaf(cv, whp[k], ah);
        ac = fmaf(cv, wcp[k], ac);
    }
    ah += bh[n]; ac += bc[n];
    h0[idx] = ah; c0[idx] = ac; hbf0[idx] = f2bf(ah);
    if (idx < B_) xtok[idx] = 1;                             // START_TOKEN
}

// ---------------- per-step: gates GEMM (WMMA bf16, LDS-staged A) + fused LSTM cell ----------------
// grid = 32 (2 hidden n-tiles each), block = 256 (8 waves: 4 gates x 2 n-subtiles)
// A = [embed(x) | h] staged ONCE per block into LDS; each wave computes both batch halves
// from one B-tile load (weights stream from L2, A fragments come from ds_load_b128).
__global__ void __launch_bounds__(256)
lstm_step_gates(const float* __restrict__ embed,              // [V,E] f32
                const unsigned short* __restrict__ wih,       // [4H,E] bf16
                const unsigned short* __restrict__ whh,       // [4H,H] bf16
                const float* __restrict__ b_ih, const float* __restrict__ b_hh,
                const int* __restrict__ xtok,
                const float* __restrict__ h_in, const float* __restrict__ c_in,
                const unsigned short* __restrict__ hbf_in,
                float* __restrict__ h_out, float* __restrict__ c_out,
                unsigned short* __restrict__ hbf_out,
                float* __restrict__ out_h, float* __restrict__ out_c,  // [B,H,T] slices
                int t) {
    __shared__ unsigned short sA[B_][SA_STRIDE];              // 32 x (512 emb | 1024 h) bf16, ~97 KB
    __shared__ float sg[2][4][B_][17];                        // gate tiles, ~17 KB

    const int tid  = threadIdx.x;
    const int lane = tid & 31;
    const int wave = tid >> 5;
    const int g    = wave & 3;                                // gate i/f/g/o
    const int nsub = wave >> 2;                               // 0/1
    const int nt   = blockIdx.x * 2 + nsub;                   // hidden tile 0..63
    const int col  = lane & 15;
    const int kh   = lane >> 4;                               // K half select (ISA A/B layout)
    const int wrow = g * H_ + nt * 16 + col;                  // weight row / output column

    // ---- stage A into LDS: gathered embedding rows (f32->bf16) ----
    for (int q = tid; q < B_ * (E_ / 4); q += 256) {          // 4096 quads, uniform
        const int b = q >> 7;                                 // E_/4 = 128
        const int k = (q & 127) * 4;
        const float4 f = *reinterpret_cast<const float4*>(embed + (size_t)xtok[b] * E_ + k);
        unsigned short* d = &sA[b][k];
        d[0] = f2bf(f.x); d[1] = f2bf(f.y); d[2] = f2bf(f.z); d[3] = f2bf(f.w);
    }
    // ---- stage A into LDS: h rows (bf16 copy, 16B chunks) ----
    for (int c = tid; c < B_ * (H_ / 8); c += 256) {          // 4096 chunks, uniform
        const int b = c >> 7;                                 // H_/8 = 128
        const int j = (c & 127) * 8;
        *reinterpret_cast<uint4*>(&sA[b][E_ + j]) =
            *reinterpret_cast<const uint4*>(hbf_in + (size_t)b * H_ + j);
    }
    __syncthreads();

    const unsigned short* sr0 = &sA[col][0];                  // batch row m=0 tile
    const unsigned short* sr1 = &sA[16 + col][0];             // batch row m=1 tile

    v8f acc0 = {};
    v8f acc1 = {};

    // phase 1: embedding contribution, K = 512 (A from LDS)
    const unsigned short* bp1 = wih + (size_t)wrow * E_;
    for (int kk = 0; kk < E_ / 32; ++kk) {
        const int ka = kk * 32 + kh * 8;
        const v8bf lo0 = *reinterpret_cast<const v8bf*>(sr0 + ka);
        const v8bf hi0 = *reinterpret_cast<const v8bf*>(sr0 + ka + 16);
        const v8bf lo1 = *reinterpret_cast<const v8bf*>(sr1 + ka);
        const v8bf hi1 = *reinterpret_cast<const v8bf*>(sr1 + ka + 16);
        v16bf a0, a1;
#pragma unroll
        for (int i = 0; i < 8; ++i) {
            a0[i] = lo0[i]; a0[i + 8] = hi0[i];
            a1[i] = lo1[i]; a1[i + 8] = hi1[i];
        }
        const v16bf bm = __builtin_shufflevector(*reinterpret_cast<const v8bf*>(bp1 + kk * 32 + kh * 8), *reinterpret_cast<const v8bf*>(bp1 + kk * 32 + kh * 8 + 16), 0, 1, 2, 3, 4, 5, 6, 7, 8, 9, 10, 11, 12, 13, 14, 15);
        acc0 = __builtin_amdgcn_wmma_f32_16x16x32_bf16(false, a0, false, bm, (short)0, acc0, false, false);
        acc1 = __builtin_amdgcn_wmma_f32_16x16x32_bf16(false, a1, false, bm, (short)0, acc1, false, false);
    }

    // phase 2: recurrent contribution, K = 1024 (A from LDS at offset E_)
    const unsigned short* bp2 = whh + (size_t)wrow * H_;
    for (int kk = 0; kk < H_ / 32; ++kk) {
        const int ka = E_ + kk * 32 + kh * 8;
        const v8bf lo0 = *reinterpret_cast<const v8bf*>(sr0 + ka);
        const v8bf hi0 = *reinterpret_cast<const v8bf*>(sr0 + ka + 16);
        const v8bf lo1 = *reinterpret_cast<const v8bf*>(sr1 + ka);
        const v8bf hi1 = *reinterpret_cast<const v8bf*>(sr1 + ka + 16);
        v16bf a0, a1;
#pragma unroll
        for (int i = 0; i < 8; ++i) {
            a0[i] = lo0[i]; a0[i + 8] = hi0[i];
            a1[i] = lo1[i]; a1[i + 8] = hi1[i];
        }
        const v16bf bm = __builtin_shufflevector(*reinterpret_cast<const v8bf*>(bp2 + kk * 32 + kh * 8), *reinterpret_cast<const v8bf*>(bp2 + kk * 32 + kh * 8 + 16), 0, 1, 2, 3, 4, 5, 6, 7, 8, 9, 10, 11, 12, 13, 14, 15);
        acc0 = __builtin_amdgcn_wmma_f32_16x16x32_bf16(false, a0, false, bm, (short)0, acc0, false, false);
        acc1 = __builtin_amdgcn_wmma_f32_16x16x32_bf16(false, a1, false, bm, (short)0, acc1, false, false);
    }

    const float bias = b_ih[wrow] + b_hh[wrow];
#pragma unroll
    for (int r = 0; r < 8; ++r) {                             // C/D layout: lanes<16 -> M=r, else M=r+8
        const int orow = (lane < 16) ? r : (r + 8);
        sg[nsub][g][orow][col]      = acc0[r] + bias;
        sg[nsub][g][orow + 16][col] = acc1[r] + bias;
    }
    __syncthreads();

    // fused LSTM cell for both hidden tiles (1024 elements, 256 threads x4)
    for (int idx = tid; idx < 2 * B_ * 16; idx += blockDim.x) {
        const int ns = idx >> 9;
        const int b  = (idx >> 4) & 31;
        const int j  = idx & 15;
        const int n  = (blockIdx.x * 2 + ns) * 16 + j;
        const float iv = sg[ns][0][b][j], fv = sg[ns][1][b][j];
        const float gv = sg[ns][2][b][j], ov = sg[ns][3][b][j];
        const size_t sidx = (size_t)b * H_ + n;
        const float hprev = h_in[sidx], cprev = c_in[sidx];
        const size_t oidx = ((size_t)b * H_ + n) * T_ + t;    // [B,H,T]
        out_h[oidx] = hprev;
        out_c[oidx] = cprev;
        const float cn = sigf(fv) * cprev + sigf(iv) * tanhf(gv);
        const float hn = sigf(ov) * tanhf(cn);
        c_out[sidx] = cn;
        h_out[sidx] = hn;
        hbf_out[sidx] = f2bf(hn);
    }
}

// ---------------- per-step: logits = h_new @ W_logits.T + b  (WMMA, LDS-staged A) ----------------
// grid = 250 (8 n-tiles each), block = 256 (8 waves, 1 n-tile each, BOTH M-tiles per wave)
// h staged once per block into LDS (64 KB); W_logits (bf16, 65 MB) streams from L2.
__global__ void __launch_bounds__(256)
logits_gemm(const unsigned short* __restrict__ wlog,          // [V,H] bf16
            const unsigned short* __restrict__ hbf,           // [B,H] bf16
            const float* __restrict__ b_log,
            float* __restrict__ logits) {                     // [B,V] f32
    __shared__ unsigned short sH[B_][SH_STRIDE];              // ~66 KB

    const int tid  = threadIdx.x;
    const int lane = tid & 31;
    const int wave = tid >> 5;
    const int nt   = blockIdx.x * 8 + wave;                   // 0..1999
    const int col  = lane & 15;
    const int kh   = lane >> 4;
    const int n    = nt * 16 + col;

    for (int c = tid; c < B_ * (H_ / 8); c += 256) {          // 4096 chunks, uniform
        const int b = c >> 7;
        const int j = (c & 127) * 8;
        *reinterpret_cast<uint4*>(&sH[b][j]) =
            *reinterpret_cast<const uint4*>(hbf + (size_t)b * H_ + j);
    }
    __syncthreads();

    const unsigned short* sr0 = &sH[col][0];
    const unsigned short* sr1 = &sH[16 + col][0];
    const unsigned short* bp  = wlog + (size_t)n * H_;

    v8f acc0 = {};
    v8f acc1 = {};
    for (int kk = 0; kk < H_ / 32; ++kk) {
        const int ka = kk * 32 + kh * 8;
        const v8bf lo0 = *reinterpret_cast<const v8bf*>(sr0 + ka);
        const v8bf hi0 = *reinterpret_cast<const v8bf*>(sr0 + ka + 16);
        const v8bf lo1 = *reinterpret_cast<const v8bf*>(sr1 + ka);
        const v8bf hi1 = *reinterpret_cast<const v8bf*>(sr1 + ka + 16);
        v16bf a0, a1;
#pragma unroll
        for (int i = 0; i < 8; ++i) {
            a0[i] = lo0[i]; a0[i + 8] = hi0[i];
            a1[i] = lo1[i]; a1[i + 8] = hi1[i];
        }
        const v16bf bm = __builtin_shufflevector(*reinterpret_cast<const v8bf*>(bp + kk * 32 + kh * 8), *reinterpret_cast<const v8bf*>(bp + kk * 32 + kh * 8 + 16), 0, 1, 2, 3, 4, 5, 6, 7, 8, 9, 10, 11, 12, 13, 14, 15);
        if (kk + 1 < H_ / 32)
            __builtin_prefetch(bp + (kk + 1) * 32 + kh * 16, 0, 1);   // global_prefetch_b8
        acc0 = __builtin_amdgcn_wmma_f32_16x16x32_bf16(false, a0, false, bm, (short)0, acc0, false, false);
        acc1 = __builtin_amdgcn_wmma_f32_16x16x32_bf16(false, a1, false, bm, (short)0, acc1, false, false);
    }
    const float bias = b_log[n];
#pragma unroll
    for (int r = 0; r < 8; ++r) {
        const int orow = (lane < 16) ? r : (r + 8);
        logits[(size_t)orow * V_ + n]        = acc0[r] + bias;
        logits[(size_t)(orow + 16) * V_ + n] = acc1[r] + bias;
    }
}

// ---------------- per-step: log-softmax + deterministic Gumbel-argmax sample ----------------
// grid = 32 (one block per batch row), block = 256
__global__ void __launch_bounds__(256)
softmax_sample(const float* __restrict__ logits,              // [B,V]
               int* __restrict__ xtok,
               float* __restrict__ out_cap,                   // [B,T]
               float* __restrict__ out_logp,                  // [B,V,T]
               int t) {
    __shared__ float smx[256], ssm[256], sbg[256];
    __shared__ int   sbi[256];
    __shared__ float sLse;
    const int b = blockIdx.x, tid = threadIdx.x;
    const float* lp = logits + (size_t)b * V_;

    float mx = -INFINITY, sm = 0.0f, bg = -INFINITY;
    int bi = 0;
    for (int v = tid; v < V_; v += 256) {
        const float x = lp[v];
        if (x > mx) { sm = sm * __expf(mx - x) + 1.0f; mx = x; }
        else        { sm += __expf(x - mx); }
        unsigned h = (unsigned)(b * 0x85EBCA6Bu) ^ (unsigned)(t * 0xC2B2AE35u) ^ (unsigned)(v * 0x27D4EB2Fu);
        h ^= h >> 16; h *= 0x7FEB352Du; h ^= h >> 15; h *= 0x846CA68Bu; h ^= h >> 16;
        const float u   = ((h >> 8) + 1u) * (1.0f / 16777217.0f);     // (0,1]
        const float gum = x - __logf(-__logf(u));
        if (gum > bg) { bg = gum; bi = v; }
    }
    smx[tid] = mx; ssm[tid] = sm; sbg[tid] = bg; sbi[tid] = bi;
    __syncthreads();
    for (int s = 128; s > 0; s >>= 1) {
        if (tid < s) {
            const float ma = smx[tid], mb = smx[tid + s];
            const float sa = ssm[tid], sb = ssm[tid + s];
            if (ma >= mb) { ssm[tid] = sa + sb * __expf(mb - ma); }
            else          { smx[tid] = mb; ssm[tid] = sb + sa * __expf(ma - mb); }
            if (sbg[tid + s] > sbg[tid]) { sbg[tid] = sbg[tid + s]; sbi[tid] = sbi[tid + s]; }
        }
        __syncthreads();
    }
    if (tid == 0) {
        sLse = smx[0] + __logf(ssm[0]);
        const int token = sbi[0];
        xtok[b] = token;
        out_cap[b * T_ + t] = (float)token;
    }
    __syncthreads();
    const float lse = sLse;
    for (int v = tid; v < V_; v += 256)
        out_logp[((size_t)b * V_ + v) * T_ + t] = lp[v] - lse;
}

// ---------------------------------------------------------------------------
extern "C" void kernel_launch(void* const* d_in, const int* in_sizes, int n_in,
                              void* d_out, int out_size, void* d_ws, size_t ws_size,
                              hipStream_t stream) {
    const float* cond  = (const float*)d_in[0];
    const float* W_h   = (const float*)d_in[1];
    const float* b_h   = (const float*)d_in[2];
    const float* W_c   = (const float*)d_in[3];
    const float* b_c   = (const float*)d_in[4];
    const float* embed = (const float*)d_in[5];
    const float* W_ih  = (const float*)d_in[6];
    const float* W_hh  = (const float*)d_in[7];
    const float* b_ih  = (const float*)d_in[8];
    const float* b_hh  = (const float*)d_in[9];
    const float* W_log = (const float*)d_in[10];
    const float* b_log = (const float*)d_in[11];
    float* out = (float*)d_out;

    // workspace carve-out (~83 MB)
    char* ws = (char*)d_ws;
    size_t off = 0;
    auto carve = [&](size_t bytes) -> char* {
        char* p = ws + off;
        off += (bytes + 255) & ~(size_t)255;
        return p;
    };
    unsigned short* wih_bf  = (unsigned short*)carve((size_t)4 * H_ * E_ * 2);
    unsigned short* whh_bf  = (unsigned short*)carve((size_t)4 * H_ * H_ * 2);
    unsigned short* wlog_bf = (unsigned short*)carve((size_t)V_ * H_ * 2);
    float* hbuf[2] = {(float*)carve((size_t)B_ * H_ * 4), (float*)carve((size_t)B_ * H_ * 4)};
    float* cbuf[2] = {(float*)carve((size_t)B_ * H_ * 4), (float*)carve((size_t)B_ * H_ * 4)};
    unsigned short* hbf[2] = {(unsigned short*)carve((size_t)B_ * H_ * 2),
                              (unsigned short*)carve((size_t)B_ * H_ * 2)};
    float* logits_ws = (float*)carve((size_t)B_ * V_ * 4);
    int*   xtok      = (int*)carve(B_ * sizeof(int));

    // output slices: captions[B,T] | logp[B,V,T] | h[B,H,T] | c[B,H,T]
    float* out_cap  = out;
    float* out_logp = out + (size_t)B_ * T_;
    float* out_h    = out_logp + (size_t)B_ * V_ * T_;
    float* out_c    = out_h + (size_t)B_ * H_ * T_;

    // 1) one-time bf16 weight conversion (W_logits becomes 65 MB -> L2 resident)
    cvt_bf16<<<512,  256, 0, stream>>>(W_ih,  wih_bf,  4 * H_ * E_);
    cvt_bf16<<<1024, 256, 0, stream>>>(W_hh,  whh_bf,  4 * H_ * H_);
    cvt_bf16<<<2048, 256, 0, stream>>>(W_log, wlog_bf, V_ * H_);

    // 2) initial state
    init_state<<<(B_ * H_) / 256, 256, 0, stream>>>(cond, W_h, b_h, W_c, b_c,
                                                    hbuf[0], cbuf[0], hbf[0], xtok);

    // 3) 48 dependent decode steps (ping-pong h/c to avoid inter-WG races)
    for (int t = 0; t < T_; ++t) {
        const int pi = t & 1, po = pi ^ 1;
        lstm_step_gates<<<H_ / 32, 256, 0, stream>>>(
            embed, wih_bf, whh_bf, b_ih, b_hh, xtok,
            hbuf[pi], cbuf[pi], hbf[pi],
            hbuf[po], cbuf[po], hbf[po],
            out_h, out_c, t);
        logits_gemm<<<V_ / 128, 256, 0, stream>>>(wlog_bf, hbf[po], b_log, logits_ws);
        softmax_sample<<<B_, 256, 0, stream>>>(logits_ws, xtok, out_cap, out_logp, t);
    }
}
